// GCNModel_66932770341395
// MI455X (gfx1250) — matrix operand order repaired and checked by reading, not compiled, not run on hardware
//
#include <hip/hip_runtime.h>
#include <hip/hip_bf16.h>

typedef __attribute__((ext_vector_type(16))) _Float16 v16h;
typedef __attribute__((ext_vector_type(8)))  float    v8f;
typedef __attribute__((ext_vector_type(4)))  int      v4i;

#define Hdim 128          // hidden width (hard requirement of the WMMA GEMM path)
#define BN_EPS 1e-5f

#define GLOBAL_AS __attribute__((address_space(1)))
#define LDS_AS    __attribute__((address_space(3)))

#if __has_builtin(__builtin_amdgcn_global_load_async_to_lds_b128)
#define HAVE_ASYNC_LDS 1
#else
#define HAVE_ASYNC_LDS 0
#endif

__device__ __forceinline__ void atomicAddF(float* p, float v) {
#if defined(__AMDGCN__)
    unsafeAtomicAdd(p, v);          // lowers to global_atomic_add_f32
#else
    atomicAdd(p, v);
#endif
}

// ---------------------------------------------------------------------------
// Pack a 128x128 f32 weight (K-major rows, N columns) into WMMA B-fragment
// order: frag f = kt*8+nt (kt 0..3, nt 0..7); per lane l, half i:
//   K = kt*32 + ((i<8)?i:i+8) + ((l>=16)?8:0),  N = nt*16 + (l&15)
// packed[mat*16384 + f*512 + l*16 + i]
// ---------------------------------------------------------------------------
__global__ void pack_weights_kernel(const float* __restrict__ W_enc,
                                    const float* __restrict__ W,
                                    _Float16* __restrict__ out, int nmat) {
    int tid = blockIdx.x * blockDim.x + threadIdx.x;
    if (tid >= nmat * Hdim * Hdim) return;
    int m   = tid >> 14;            // /16384
    int rem = tid & 16383;
    int f   = rem >> 9;             // fragment 0..31
    int l   = (rem >> 4) & 31;      // lane
    int i   = rem & 15;             // half index
    int kt  = f >> 3, nt = f & 7;
    int K   = kt * 32 + ((i < 8) ? i : i + 8) + ((l >= 16) ? 8 : 0);
    int Nc  = nt * 16 + (l & 15);
    const float* src = (m == 0) ? W_enc : (W + (size_t)(m - 1) * Hdim * Hdim);
    out[tid] = (_Float16)src[K * Hdim + Nc];
}

// deg=1 (self loop), pool=0, cnt=0
__global__ void init_misc_kernel(float* deg, float* pool, float* cnt,
                                 int N, int BH, int B) {
    int i = blockIdx.x * blockDim.x + threadIdx.x;
    if (i < N)  deg[i]  = 1.0f;
    if (i < BH) pool[i] = 0.0f;
    if (i < B)  cnt[i]  = 0.0f;
}

__global__ void degree_kernel(const long long* __restrict__ col,
                              float* __restrict__ deg, int E) {
    int e = blockIdx.x * blockDim.x + threadIdx.x;
    if (e < E) atomicAddF(&deg[(int)col[e]], 1.0f);
}

__global__ void rsqrt_kernel(float* deg, int N) {
    int i = blockIdx.x * blockDim.x + threadIdx.x;
    if (i < N) { float d = deg[i]; deg[i] = (d > 0.0f) ? rsqrtf(d) : 0.0f; }
}

// ---------------------------------------------------------------------------
// C[M x 128] = f16(A[M x 128]) @ Bpacked (+ bias).  One 16-row tile per wave,
// 8 waves per block.  The packed weight matrix (32 KB) is staged into LDS
// once per block via gfx1250 async global->LDS copies; per K-step all eight
// B fragments are fetched into distinct registers first so the ds_load
// latency overlaps the WMMA burst instead of serializing on s_wait_dscnt 0.
// M must be a multiple of 16 (50000 = 3125*16).
// ---------------------------------------------------------------------------
__global__ __launch_bounds__(256)
void gemm_wmma_kernel(const float* __restrict__ A, const _Float16* __restrict__ Bp,
                      const float* __restrict__ bias, float* __restrict__ C, int M) {
    __shared__ __align__(32) _Float16 bsh[Hdim * Hdim];   // 32 KB: full packed weight

    const int tid = threadIdx.x;
    // ---- cooperative B staging: 2048 x 16B chunks, 8 per thread ----
#if HAVE_ASYNC_LDS
#pragma unroll
    for (int it = 0; it < 8; ++it) {
        int idx = (it * 256 + tid) * 8;               // offset in halves (16 B chunks)
        __builtin_amdgcn_global_load_async_to_lds_b128(
            (GLOBAL_AS v4i*)(Bp + idx),
            (LDS_AS v4i*)(bsh + idx), 0, 0);
    }
#if __has_builtin(__builtin_amdgcn_s_wait_asynccnt)
    __builtin_amdgcn_s_wait_asynccnt(0);
#else
    asm volatile("s_wait_asynccnt 0" ::: "memory");
#endif
#else
#pragma unroll
    for (int it = 0; it < 8; ++it) {
        int idx = (it * 256 + tid) * 8;
        *(int4*)(bsh + idx) = *(const int4*)(Bp + idx);
    }
#endif
    __syncthreads();

    const int lane  = tid & 31;
    const int wave  = tid >> 5;
    const int mtile = blockIdx.x * 8 + wave;
    if (mtile * 16 >= M) return;               // wave-uniform: EXEC stays all-ones
    const int m0    = mtile * 16;
    const int lhalf = (lane >= 16) ? 1 : 0;
    const int l15   = lane & 15;

    v8f acc[8];
#pragma unroll
    for (int nt = 0; nt < 8; ++nt) {
        float bv = bias ? bias[nt * 16 + l15] : 0.0f;
#pragma unroll
        for (int r = 0; r < 8; ++r) acc[nt][r] = bv;
    }

    const float* arow = A + (size_t)(m0 + l15) * Hdim;
#pragma unroll
    for (int kt = 0; kt < 4; ++kt) {
        // A frag: lane holds row m0+l15; halves 0..7 = K kb..kb+7, 8..15 = K kb+16..kb+23
        const int kb = kt * 32 + lhalf * 8;
        float4 c0a = *(const float4*)(arow + kb);
        float4 c0b = *(const float4*)(arow + kb + 4);
        float4 c1a = *(const float4*)(arow + kb + 16);
        float4 c1b = *(const float4*)(arow + kb + 20);
        v16h a;
        a[0]=(_Float16)c0a.x; a[1]=(_Float16)c0a.y; a[2]=(_Float16)c0a.z; a[3]=(_Float16)c0a.w;
        a[4]=(_Float16)c0b.x; a[5]=(_Float16)c0b.y; a[6]=(_Float16)c0b.z; a[7]=(_Float16)c0b.w;
        a[8]=(_Float16)c1a.x; a[9]=(_Float16)c1a.y; a[10]=(_Float16)c1a.z; a[11]=(_Float16)c1a.w;
        a[12]=(_Float16)c1b.x; a[13]=(_Float16)c1b.y; a[14]=(_Float16)c1b.z; a[15]=(_Float16)c1b.w;

        // hoist all eight B fragments for this K-step into registers
        v16h bf[8];
#pragma unroll
        for (int nt = 0; nt < 8; ++nt)
            bf[nt] = *(const v16h*)(bsh + ((kt * 8 + nt) * 512 + lane * 16));
#pragma unroll
        for (int nt = 0; nt < 8; ++nt)
            acc[nt] = __builtin_amdgcn_wmma_f32_16x16x32_f16(
                false, a, false, bf[nt], (short)0, acc[nt], false, false);
    }
#pragma unroll
    for (int nt = 0; nt < 8; ++nt) {
#pragma unroll
        for (int r = 0; r < 8; ++r) {
            int row = m0 + r + lhalf * 8;
            C[(size_t)row * Hdim + nt * 16 + l15] = acc[nt][r];
        }
    }
}

// agg[r][c] = bias[c]; block 0 also zeroes the BN stats accumulators
__global__ void init_agg_kernel(float* __restrict__ agg, const float* __restrict__ bias,
                                float* __restrict__ stats, int total) {
    int i = blockIdx.x * blockDim.x + threadIdx.x;
    if (blockIdx.x == 0 && threadIdx.x < 2 * Hdim) stats[threadIdx.x] = 0.0f;
    if (i < total) agg[i] = bias[i & (Hdim - 1)];
}

// one wave per edge (incl. N virtual self loops); lane handles 4 features
__global__ __launch_bounds__(256)
void scatter_kernel(const float* __restrict__ t, const long long* __restrict__ rowi,
                    const long long* __restrict__ coli, const float* __restrict__ dinv,
                    float* __restrict__ agg, int E, int N) {
    int gid  = blockIdx.x * blockDim.x + threadIdx.x;
    int edge = gid >> 5;
    int lane = gid & 31;
    if (edge >= E + N) return;
    int src, dst;
    if (edge < E) { src = (int)rowi[edge]; dst = (int)coli[edge]; }
    else          { src = dst = edge - E; }
    float nrm = dinv[src] * dinv[dst];
    const float4 v = *(const float4*)(t + (size_t)src * Hdim + lane * 4);
    float* ap = agg + (size_t)dst * Hdim + lane * 4;
    atomicAddF(ap + 0, nrm * v.x);
    atomicAddF(ap + 1, nrm * v.y);
    atomicAddF(ap + 2, nrm * v.z);
    atomicAddF(ap + 3, nrm * v.w);
}

// per-channel sum / sumsq (coalesced: 256 threads cover 2 rows x 128 channels)
__global__ __launch_bounds__(256)
void stats_kernel(const float* __restrict__ agg, float* __restrict__ stats, int N) {
    int c    = threadIdx.x & (Hdim - 1);
    int half = threadIdx.x >> 7;
    float s = 0.0f, sq = 0.0f;
    for (int r = blockIdx.x * 2 + half; r < N; r += gridDim.x * 2) {
        float v = agg[(size_t)r * Hdim + c];
        s += v; sq += v * v;
    }
    atomicAddF(&stats[c], s);
    atomicAddF(&stats[Hdim + c], sq);
}

__global__ void bnparam_kernel(const float* __restrict__ stats,
                               const float* __restrict__ gamma,
                               const float* __restrict__ beta,
                               float* __restrict__ sc, int N) {
    int c = threadIdx.x;
    if (c >= Hdim) return;
    float invN  = 1.0f / (float)N;
    float mean  = stats[c] * invN;
    float var   = stats[Hdim + c] * invN - mean * mean;
    float scale = gamma[c] * rsqrtf(var + BN_EPS);
    sc[c]        = scale;
    sc[Hdim + c] = beta[c] - mean * scale;
}

__global__ void bnrelu_kernel(const float* __restrict__ agg, const float* __restrict__ sc,
                              float* __restrict__ h, int total) {
    int i = blockIdx.x * blockDim.x + threadIdx.x;
    if (i >= total) return;
    int c = i & (Hdim - 1);
    float v = fmaf(agg[i], sc[c], sc[Hdim + c]);
    h[i] = fmaxf(v, 0.0f);
}

__global__ __launch_bounds__(256)
void pool_kernel(const float* __restrict__ h, const long long* __restrict__ batch,
                 float* __restrict__ pool, float* __restrict__ cnt, int N) {
    int gid  = blockIdx.x * blockDim.x + threadIdx.x;
    int row  = gid >> 5;
    int lane = gid & 31;
    if (row >= N) return;
    int b = (int)batch[row];
    const float4 v = *(const float4*)(h + (size_t)row * Hdim + lane * 4);
    float* pp = pool + (size_t)b * Hdim + lane * 4;
    atomicAddF(pp + 0, v.x);
    atomicAddF(pp + 1, v.y);
    atomicAddF(pp + 2, v.z);
    atomicAddF(pp + 3, v.w);
    if (lane == 0) atomicAddF(&cnt[b], 1.0f);
}

__global__ void cls_kernel(const float* __restrict__ pool, const float* __restrict__ cnt,
                           const float* __restrict__ Wc, const float* __restrict__ bc,
                           float* __restrict__ out, int B, int OUT) {
    int tid = blockIdx.x * blockDim.x + threadIdx.x;
    if (tid >= B * OUT) return;
    int b = tid / OUT, o = tid - b * OUT;
    float inv = 1.0f / fmaxf(cnt[b], 1.0f);
    float acc = 0.0f;
#pragma unroll 8
    for (int k = 0; k < Hdim; ++k) acc = fmaf(pool[(size_t)b * Hdim + k], Wc[k * OUT + o], acc);
    out[tid] = acc * inv + bc[o];
}

static inline size_t align256(size_t x) { return (x + 255) & ~(size_t)255; }

extern "C" void kernel_launch(void* const* d_in, const int* in_sizes, int n_in,
                              void* d_out, int out_size, void* d_ws, size_t ws_size,
                              hipStream_t stream) {
    const float*     x      = (const float*)d_in[0];
    const long long* ei     = (const long long*)d_in[1];   // int64 per reference
    const long long* batch  = (const long long*)d_in[2];
    const float*     W_enc  = (const float*)d_in[3];
    const float*     b_enc  = (const float*)d_in[4];
    const float*     W      = (const float*)d_in[5];
    const float*     b      = (const float*)d_in[6];
    const float*     gamma  = (const float*)d_in[7];
    const float*     beta   = (const float*)d_in[8];
    const float*     W_cls  = (const float*)d_in[9];
    const float*     b_cls  = (const float*)d_in[10];
    float*           out    = (float*)d_out;

    const int H   = in_sizes[4];            // 128
    const int IN  = in_sizes[3] / H;        // 128
    const int N   = in_sizes[0] / IN;       // 50000
    const int E   = in_sizes[1] / 2;        // 600000
    const int L   = in_sizes[6] / H;        // 4
    const int OUT = in_sizes[10];           // 10
    const int B   = out_size / OUT;         // 256
    const int total = N * H;

    // workspace carve-up
    char* ws = (char*)d_ws;
    size_t off = 0;
    float* deg   = (float*)(ws + off); off = align256(off + (size_t)N * 4);
    float* stats = (float*)(ws + off); off = align256(off + 2 * H * 4);
    float* sc    = (float*)(ws + off); off = align256(off + 2 * H * 4);
    float* pool  = (float*)(ws + off); off = align256(off + (size_t)B * H * 4);
    float* cnt   = (float*)(ws + off); off = align256(off + (size_t)B * 4);
    _Float16* wpack = (_Float16*)(ws + off); off = align256(off + (size_t)(L + 1) * H * H * 2);
    float* h   = (float*)(ws + off); off = align256(off + (size_t)total * 4);
    float* t   = (float*)(ws + off); off = align256(off + (size_t)total * 4);
    float* agg = (float*)(ws + off); off = align256(off + (size_t)total * 4);
    (void)ws_size; (void)n_in;

    const long long* row = ei;        // edge_index[0]
    const long long* col = ei + E;    // edge_index[1]
    const int TB = 256;

    // 1) pack weights into WMMA B-fragment order (encoder + L layers)
    int packN = (L + 1) * H * H;
    pack_weights_kernel<<<(packN + TB - 1) / TB, TB, 0, stream>>>(W_enc, W, wpack, L + 1);

    // 2) deg = 1 (self loop) + scatter count; pool/cnt zeroed
    int initN = max(N, B * H);
    init_misc_kernel<<<(initN + TB - 1) / TB, TB, 0, stream>>>(deg, pool, cnt, N, B * H, B);
    degree_kernel<<<(E + TB - 1) / TB, TB, 0, stream>>>(col, deg, E);
    rsqrt_kernel<<<(N + TB - 1) / TB, TB, 0, stream>>>(deg, N);   // deg -> d^-1/2 in place

    const int mtiles = (N + 15) / 16;
    const int gemmBlocks = (mtiles + 7) / 8;

    // 3) encoder: h = x @ W_enc + b_enc
    gemm_wmma_kernel<<<gemmBlocks, TB, 0, stream>>>(x, wpack, b_enc, h, N);

    // 4) GCN layers
    const int EE = E + N;
    for (int l = 0; l < L; ++l) {
        gemm_wmma_kernel<<<gemmBlocks, TB, 0, stream>>>(
            h, wpack + (size_t)(l + 1) * H * H, nullptr, t, N);
        init_agg_kernel<<<(total + TB - 1) / TB, TB, 0, stream>>>(agg, b + l * H, stats, total);
        scatter_kernel<<<((size_t)EE * 32 + TB - 1) / TB, TB, 0, stream>>>(
            t, row, col, deg, agg, E, N);
        stats_kernel<<<1024, TB, 0, stream>>>(agg, stats, N);
        bnparam_kernel<<<1, H, 0, stream>>>(stats, gamma + l * H, beta + l * H, sc, N);
        bnrelu_kernel<<<(total + TB - 1) / TB, TB, 0, stream>>>(agg, sc, h, total);
    }

    // 5) global mean pool + classifier
    pool_kernel<<<((size_t)N * 32 + TB - 1) / TB, TB, 0, stream>>>(h, batch, pool, cnt, N);
    cls_kernel<<<(B * OUT + TB - 1) / TB, TB, 0, stream>>>(pool, cnt, W_cls, b_cls, out, B, OUT);
}
